// GraphTransformerDenoisingNetwork_8005819040032
// MI455X (gfx1250) — compile-verified, matrix operand order checked by reading, not run on hardware
//
#include <hip/hip_runtime.h>
#include <hip/hip_bf16.h>

// ---------------------------------------------------------------------------
// GraphTransformerDenoisingNetwork forward for MI455X (gfx1250, wave32).
// All heavy GEMMs use v_wmma_f32_16x16x32_bf16 (bf16 in, f32 accumulate).
// Weights are pre-swizzled into per-lane WMMA fragment order so every B
// fragment is ONE contiguous 32B load (strength-reduced pointer stepping);
// A fragments are two 16B vector loads.  Edge-side work (512^2-row tensors)
// is fused in LDS so the 256MB YE intermediate is never materialized;
// inter-kernel edge buffers are bf16 so the ~80MB working set stays in the
// 192MB L2.
// ---------------------------------------------------------------------------

#define NN   512
#define HH   8
#define DQK  32

typedef __attribute__((ext_vector_type(16))) __bf16 v16bf;
typedef __attribute__((ext_vector_type(8)))  __bf16 v8bf;
typedef __attribute__((ext_vector_type(8)))  float  v8f;

// ---- WMMA fragment helpers (ISA 7.12.2 layouts, wave32) -------------------
// A tile 16x32 bf16: lane&15 = row M, (lane>>4)*8 = K base.  Per lane the
// fragment is two contiguous 8-element runs: K = kb+[0..7] and kb+16+[0..7].
__device__ __forceinline__ v16bf frag_a(const __bf16* A, int lda, int lane) {
  const __bf16* p = A + (lane & 15) * lda + ((lane >> 4) << 3);
  v8bf lo = *(const v8bf*)p;
  v8bf hi = *(const v8bf*)(p + 16);
  return __builtin_shufflevector(lo, hi, 0, 1, 2, 3, 4, 5, 6, 7,
                                 8, 9, 10, 11, 12, 13, 14, 15);
}

// B fragments come from fragment-packed weights: [tile][lane][16 bf16].
__device__ __forceinline__ v16bf frag_ld(const __bf16* p) {
  return *(const v16bf*)p;
}
__device__ __forceinline__ const __bf16* frag_bp(const __bf16* B, int nt, int lane) {
  return B + (((long)nt << 5) + lane) * 16;   // start at kt=0 for column-tile nt
}
// element stride per kt step = (#column tiles) * 32 lanes * 16 elems
#define BSTEP(nt16) ((long)(nt16) << 9)

__device__ __forceinline__ v8f wmma_bf16(v16bf a, v16bf b, v8f c) {
  return __builtin_amdgcn_wmma_f32_16x16x32_bf16(false, a, false, b,
                                                 (short)0, c, false, false);
}

// C/D 16x16 f32: N = lane&15; VGPR r -> M = r + 8*(lane>=16).
__device__ __forceinline__ void store_lds_bf16(__bf16* C, int ldc, int lane,
                                               v8f acc, const float* bias,
                                               bool relu) {
  int n = lane & 15, mo = (lane >> 4) << 3;
  float bs = bias[n];
#pragma unroll
  for (int r = 0; r < 8; ++r) {
    float x = acc[r] + bs;
    if (relu) x = fmaxf(x, 0.f);
    C[(mo + r) * ldc + n] = (__bf16)x;
  }
}

// ---- weight repack: w(dout,din) f32 -> fragment-packed bf16 ---------------
// Layout: tile = kt*(N/16)+nt ; element (tile, lane, e) holds
// B[k = kt*32 + kb(lane) + krel(e)][n = nt*16 + (lane&15)].
__global__ void pack_w(const float* __restrict__ w, int dout, int din,
                       __bf16* __restrict__ B, int K, int N) {
  int idx = blockIdx.x * 256 + threadIdx.x;
  if (idx >= K * N) return;
  int e    = idx & 15;
  int lane = (idx >> 4) & 31;
  int tile = idx >> 9;
  int nt16 = N >> 4;
  int nt = tile % nt16, kt = tile / nt16;
  int n  = nt * 16 + (lane & 15);
  int tt = e >> 1;
  int k  = kt * 32 + ((lane >> 4) << 3) +
           (((tt & 4) << 2) | ((tt & 3) << 1)) + (e & 1);
  float v = (k < din && n < dout) ? w[n * din + k] : 0.f;
  B[idx] = (__bf16)v;
}

__global__ void pack_bias(const float* __restrict__ b, int dout,
                          float* __restrict__ bp, int Np) {
  int n = blockIdx.x * 256 + threadIdx.x;
  if (n >= Np) return;
  bp[n] = (n < dout) ? b[n] : 0.f;
}

// ---- structural degree features -------------------------------------------
__global__ __launch_bounds__(256) void deg_kernel(const float* __restrict__ E,
                                                  float* __restrict__ indeg,
                                                  float* __restrict__ outdeg) {
  __shared__ float rs[256], cs[256];
  int i = blockIdx.x, t = threadIdx.x;
  float r = 0.f, c = 0.f;
  for (int j = t; j < NN; j += 256) {
    r += E[((long)i * NN + j) * 4 + 1];   // A[i][j]
    c += E[((long)j * NN + i) * 4 + 1];   // A[j][i]
  }
  rs[t] = r; cs[t] = c;
  __syncthreads();
  for (int s = 128; s; s >>= 1) {
    if (t < s) { rs[t] += rs[t + s]; cs[t] += cs[t + s]; }
    __syncthreads();
  }
  if (!t) { outdeg[i] = rs[0]; indeg[i] = cs[0]; }
}

__global__ __launch_bounds__(256) void featF_kernel(const float* __restrict__ E,
                                                    const float* __restrict__ indeg,
                                                    const float* __restrict__ outdeg,
                                                    float* __restrict__ F) {
  __shared__ float r1[256], r2[256], r3[256], r4[256];
  int n = blockIdx.x, t = threadIdx.x;
  float s1 = 0, s2 = 0, s3 = 0, s4 = 0;
  for (int m = t; m < NN; m += 256) {
    float aT = E[((long)m * NN + n) * 4 + 1];  // A[m][n]
    float aR = E[((long)n * NN + m) * 4 + 1];  // A[n][m]
    s1 += aT * indeg[m];  s2 += aT * outdeg[m];
    s3 += aR * indeg[m];  s4 += aR * outdeg[m];
  }
  r1[t] = s1; r2[t] = s2; r3[t] = s3; r4[t] = s4;
  __syncthreads();
  for (int s = 128; s; s >>= 1) {
    if (t < s) { r1[t] += r1[t+s]; r2[t] += r2[t+s]; r3[t] += r3[t+s]; r4[t] += r4[t+s]; }
    __syncthreads();
  }
  if (!t) {
    float id = indeg[n], od = outdeg[n];
    F[n*7+0] = id + od; F[n*7+1] = id; F[n*7+2] = od;
    F[n*7+3] = r1[0];   F[n*7+4] = r2[0];
    F[n*7+5] = r3[0];   F[n*7+6] = r4[0];
  }
}

__global__ void build_xin(const float* __restrict__ X, const float* __restrict__ F,
                          __bf16* __restrict__ Xin) {
  int idx = blockIdx.x * 256 + threadIdx.x;
  if (idx >= NN * 32) return;
  int r = idx >> 5, c = idx & 31;
  float v = 0.f;
  if (c < 8) v = X[r * 8 + c];
  else if (c < 15) v = F[r * 7 + (c - 8)];
  Xin[idx] = (__bf16)v;
}

// ---- generic node-level GEMM: C = act(A(MxK)bf16 @ Bfrag + bias) ----------
__global__ __launch_bounds__(256) void gemm_node(const __bf16* __restrict__ A,
                                                 int M, int K,
                                                 const __bf16* __restrict__ B,
                                                 int N, const float* __restrict__ bias,
                                                 float* __restrict__ Cf,
                                                 __bf16* __restrict__ Cb, int relu) {
  int wave = threadIdx.x >> 5, lane = threadIdx.x & 31;
  int nt16 = N >> 4;
  int ntiles = (M >> 4) * nt16;
  int tile = blockIdx.x * 8 + wave;
  if (tile >= ntiles) return;
  int mt = tile / nt16, nt = tile % nt16;
  v8f acc = {};
  const __bf16* ap = A + mt * 16 * K;
  const __bf16* bp = frag_bp(B, nt, lane);
  const long bstep = BSTEP(nt16);
  for (int kt = 0; kt < (K >> 5); ++kt) {
    acc = wmma_bf16(frag_a(ap + kt * 32, K, lane), frag_ld(bp), acc);
    bp += bstep;
  }
  int n = lane & 15, mo = (lane >> 4) << 3;
  float bs = bias[nt * 16 + n];
#pragma unroll
  for (int r = 0; r < 8; ++r) {
    float x = acc[r] + bs;
    if (relu) x = fmaxf(x, 0.f);
    int row = mt * 16 + mo + r, col = nt * 16 + n;
    if (Cf) Cf[row * N + col] = x;
    if (Cb) Cb[row * N + col] = (__bf16)x;
  }
}

// ---- fused in_E MLP: (4->256 relu ->256 relu ->64), 32 edge rows/block ----
__global__ __launch_bounds__(256) void edge_in_mlp(const float* __restrict__ E,
    const __bf16* W0, const float* b0, const __bf16* W1, const float* b1,
    const __bf16* W2, const float* b2, __bf16* __restrict__ eout) {
  __shared__ alignas(32) __bf16 s_in[32 * 32];
  __shared__ alignas(32) __bf16 s_h1[32 * 256];
  __shared__ alignas(32) __bf16 s_h2[32 * 256];
  int t = threadIdx.x, wave = t >> 5, lane = t & 31;
  long base = (long)blockIdx.x * 32;
  if (t < 32) {
    const float4 ev = *(const float4*)(E + (base + t) * 4);
    __bf16* row = s_in + t * 32;
    row[0] = (__bf16)ev.x; row[1] = (__bf16)ev.y;
    row[2] = (__bf16)ev.z; row[3] = (__bf16)ev.w;
#pragma unroll
    for (int c = 4; c < 32; ++c) row[c] = (__bf16)0.f;
  }
  __syncthreads();
  for (int tile = wave; tile < 32; tile += 8) {      // 32x32 @ 32x256
    int mt = tile >> 4, nt = tile & 15;
    v8f acc = {};
    acc = wmma_bf16(frag_a(s_in + mt * 16 * 32, 32, lane),
                    frag_ld(frag_bp(W0, nt, lane)), acc);
    store_lds_bf16(s_h1 + mt * 16 * 256 + nt * 16, 256, lane, acc, b0 + nt * 16, true);
  }
  __syncthreads();
  for (int tile = wave; tile < 32; tile += 8) {      // 32x256 @ 256x256
    int mt = tile >> 4, nt = tile & 15;
    v8f acc = {};
    const __bf16* bp = frag_bp(W1, nt, lane);
#pragma unroll
    for (int kt = 0; kt < 8; ++kt) {
      acc = wmma_bf16(frag_a(s_h1 + mt * 16 * 256 + kt * 32, 256, lane),
                      frag_ld(bp), acc);
      bp += BSTEP(16);
    }
    store_lds_bf16(s_h2 + mt * 16 * 256 + nt * 16, 256, lane, acc, b1 + nt * 16, true);
  }
  __syncthreads();
  for (int tile = wave; tile < 8; tile += 8) {       // 32x256 @ 256x64
    int mt = tile >> 2, nt = tile & 3;
    v8f acc = {};
    const __bf16* bp = frag_bp(W2, nt, lane);
#pragma unroll
    for (int kt = 0; kt < 8; ++kt) {
      acc = wmma_bf16(frag_a(s_h2 + mt * 16 * 256 + kt * 32, 256, lane),
                      frag_ld(bp), acc);
      bp += BSTEP(4);
    }
    int n = lane & 15, mo = (lane >> 4) << 3;
    float bs = b2[nt * 16 + n];
#pragma unroll
    for (int r = 0; r < 8; ++r)
      eout[(base + mt * 16 + mo + r) * 64 + nt * 16 + n] = (__bf16)(acc[r] + bs);
  }
}

// ---- fused FiLM + logits + E_out: YE never leaves LDS ---------------------
// YE = add(e) + (mul(e)+1) * (q_i (*) k_j);  logit[h] = sum_32 YE;
// e_next = YE @ W_Eout + b.
__global__ __launch_bounds__(256) void film_kernel(const __bf16* __restrict__ e,
    const float* __restrict__ q, const float* __restrict__ k,
    const __bf16* Wadd, const float* badd, const __bf16* Wmul, const float* bmul,
    const __bf16* WE, const float* bE,
    float* __restrict__ Lg, __bf16* __restrict__ enext) {
  __shared__ alignas(32) __bf16 s_e[32 * 64];
  __shared__ float  s_y[32 * 256];   // add result, then f32 YE
  __shared__ alignas(32) __bf16 s_ye[32 * 256];
  __shared__ float  s_qi[256];
  int t = threadIdx.x, wave = t >> 5, lane = t & 31;
  int i = blockIdx.y, jb = blockIdx.x * 32;
  long ebase = (long)i * NN + jb;
  s_qi[t] = q[i * 256 + t];
  ((v8bf*)s_e)[t] = ((const v8bf*)(e + ebase * 64))[t];   // 32x64 = 256 chunks
  __builtin_prefetch(k + (jb + (t >> 3)) * 256 + ((t & 7) << 5), 0, 3);
  __syncthreads();
  for (int tile = wave; tile < 32; tile += 8) {      // add: 32x64 @ 64x256
    int mt = tile >> 4, nt = tile & 15;
    v8f acc = {};
    const __bf16* bp = frag_bp(Wadd, nt, lane);
#pragma unroll
    for (int kt = 0; kt < 2; ++kt) {
      acc = wmma_bf16(frag_a(s_e + mt * 16 * 64 + kt * 32, 64, lane),
                      frag_ld(bp), acc);
      bp += BSTEP(16);
    }
    int n = lane & 15, mo = (lane >> 4) << 3;
    float bs = badd[nt * 16 + n];
#pragma unroll
    for (int r = 0; r < 8; ++r)
      s_y[(mt * 16 + mo + r) * 256 + nt * 16 + n] = acc[r] + bs;
  }
  __syncthreads();
  for (int tile = wave; tile < 32; tile += 8) {      // mul GEMM + combine
    int mt = tile >> 4, nt = tile & 15;
    v8f acc = {};
    const __bf16* bp = frag_bp(Wmul, nt, lane);
#pragma unroll
    for (int kt = 0; kt < 2; ++kt) {
      acc = wmma_bf16(frag_a(s_e + mt * 16 * 64 + kt * 32, 64, lane),
                      frag_ld(bp), acc);
      bp += BSTEP(16);
    }
    int n = lane & 15, mo = (lane >> 4) << 3;
    int nabs = nt * 16 + n;
    float qi = s_qi[nabs], bm = bmul[nabs];
#pragma unroll
    for (int r = 0; r < 8; ++r) {
      int jl = mt * 16 + mo + r;
      float kj = k[(jb + jl) * 256 + nabs];
      float ye = s_y[jl * 256 + nabs] + (acc[r] + bm + 1.0f) * (qi * kj);
      s_y[jl * 256 + nabs]  = ye;
      s_ye[jl * 256 + nabs] = (__bf16)ye;
    }
  }
  __syncthreads();
  {                                                   // per-head logits
    int jl = t >> 3, h = t & 7;
    float s = 0.f;
#pragma unroll
    for (int d = 0; d < DQK; ++d) s += s_y[jl * 256 + h * DQK + d];
    Lg[(ebase + jl) * HH + h] = s;
  }
  for (int tile = wave; tile < 8; tile += 8) {        // E_out: 32x256 @ 256x64
    int mt = tile >> 2, nt = tile & 3;
    v8f acc = {};
    const __bf16* bp = frag_bp(WE, nt, lane);
#pragma unroll
    for (int kt = 0; kt < 8; ++kt) {
      acc = wmma_bf16(frag_a(s_ye + mt * 16 * 256 + kt * 32, 256, lane),
                      frag_ld(bp), acc);
      bp += BSTEP(4);
    }
    int n = lane & 15, mo = (lane >> 4) << 3;
    float bs = bE[nt * 16 + n];
#pragma unroll
    for (int r = 0; r < 8; ++r)
      enext[(ebase + mt * 16 + mo + r) * 64 + nt * 16 + n] = (__bf16)(acc[r] + bs);
  }
}

// ---- softmax over j + weighted V aggregation (one node per block) ---------
__global__ __launch_bounds__(256) void attn_kernel(const float* __restrict__ Lg,
                                                   const float* __restrict__ v,
                                                   __bf16* __restrict__ Mout) {
  __shared__ alignas(16) float s_l[NN * HH];
  int i = blockIdx.x, t = threadIdx.x;
  for (int idx = t; idx < NN * HH / 4; idx += 256)
    ((float4*)s_l)[idx] = ((const float4*)(Lg + (long)i * NN * HH))[idx];
  __syncthreads();
  int h = t >> 5, d = t & 31;         // wave g == head g (wave32)
  float mx = -1e30f;
  for (int j = d; j < NN; j += 32) mx = fmaxf(mx, s_l[j * HH + h]);
  for (int o = 16; o; o >>= 1) mx = fmaxf(mx, __shfl_xor(mx, o, 32));
  float sum = 0.f;
  for (int j = d; j < NN; j += 32) sum += __expf(s_l[j * HH + h] - mx);
  for (int o = 16; o; o >>= 1) sum += __shfl_xor(sum, o, 32);
  float inv = 1.0f / sum;
  float acc = 0.f;
  for (int j = 0; j < NN; ++j) {
    float w = __expf(s_l[j * HH + h] - mx) * inv;
    acc += w * v[j * 256 + t];        // t == h*32+d
  }
  Mout[i * 256 + t] = (__bf16)acc;
}

// ---- fused out_E MLP (64->256 relu ->256 relu ->4) + softmax(4) -----------
__global__ __launch_bounds__(256) void edge_out_mlp(const __bf16* __restrict__ e,
    const __bf16* W0, const float* b0, const __bf16* W1, const float* b1,
    const __bf16* W2, const float* b2, float* __restrict__ Eo) {
  __shared__ alignas(32) __bf16 s_in[32 * 64];
  __shared__ alignas(32) __bf16 s_h1[32 * 256];
  __shared__ alignas(32) __bf16 s_h2[32 * 256];
  __shared__ float  s_out[32 * 16];
  int t = threadIdx.x, wave = t >> 5, lane = t & 31;
  long base = (long)blockIdx.x * 32;
  ((v8bf*)s_in)[t] = ((const v8bf*)(e + base * 64))[t];
  __syncthreads();
  for (int tile = wave; tile < 32; tile += 8) {
    int mt = tile >> 4, nt = tile & 15;
    v8f acc = {};
    const __bf16* bp = frag_bp(W0, nt, lane);
#pragma unroll
    for (int kt = 0; kt < 2; ++kt) {
      acc = wmma_bf16(frag_a(s_in + mt * 16 * 64 + kt * 32, 64, lane),
                      frag_ld(bp), acc);
      bp += BSTEP(16);
    }
    store_lds_bf16(s_h1 + mt * 16 * 256 + nt * 16, 256, lane, acc, b0 + nt * 16, true);
  }
  __syncthreads();
  for (int tile = wave; tile < 32; tile += 8) {
    int mt = tile >> 4, nt = tile & 15;
    v8f acc = {};
    const __bf16* bp = frag_bp(W1, nt, lane);
#pragma unroll
    for (int kt = 0; kt < 8; ++kt) {
      acc = wmma_bf16(frag_a(s_h1 + mt * 16 * 256 + kt * 32, 256, lane),
                      frag_ld(bp), acc);
      bp += BSTEP(16);
    }
    store_lds_bf16(s_h2 + mt * 16 * 256 + nt * 16, 256, lane, acc, b1 + nt * 16, true);
  }
  __syncthreads();
  for (int tile = wave; tile < 2; tile += 8) {        // 32x256 @ 256x16
    int mt = tile;
    v8f acc = {};
    const __bf16* bp = frag_bp(W2, 0, lane);
#pragma unroll
    for (int kt = 0; kt < 8; ++kt) {
      acc = wmma_bf16(frag_a(s_h2 + mt * 16 * 256 + kt * 32, 256, lane),
                      frag_ld(bp), acc);
      bp += BSTEP(1);
    }
    int n = lane & 15, mo = (lane >> 4) << 3;
    float bs = b2[n];
#pragma unroll
    for (int r = 0; r < 8; ++r) s_out[(mt * 16 + mo + r) * 16 + n] = acc[r] + bs;
  }
  __syncthreads();
  if (t < 32) {
    float m4 = -1e30f;
    for (int c = 0; c < 4; ++c) m4 = fmaxf(m4, s_out[t * 16 + c]);
    float sum = 0.f, ex[4];
    for (int c = 0; c < 4; ++c) { ex[c] = __expf(s_out[t * 16 + c] - m4); sum += ex[c]; }
    for (int c = 0; c < 4; ++c) Eo[(base + t) * 4 + c] = ex[c] / sum;
  }
}

__global__ void softmax_x(const float* __restrict__ Xf, float* __restrict__ out) {
  int r = blockIdx.x * blockDim.x + threadIdx.x;
  if (r >= NN) return;
  float mx = -1e30f;
  for (int c = 0; c < 8; ++c) mx = fmaxf(mx, Xf[r * 16 + c]);
  float s = 0.f, e[8];
  for (int c = 0; c < 8; ++c) { e[c] = __expf(Xf[r * 16 + c] - mx); s += e[c]; }
  for (int c = 0; c < 8; ++c) out[r * 8 + c] = e[c] / s;
}

// ---------------------------------------------------------------------------
// Host orchestration.  Input leaf order assumed: X, E, then params flattened
// jax-pytree-style (dict keys sorted; each linear contributes b then w):
//   in_E[0..2], in_X[0..2], layers[0..1]{E_out,K,Q,V,X_out,film.add,film.mul},
//   out_E[0..2], out_X[0..2]  ->  54 leaves total.
// ---------------------------------------------------------------------------
extern "C" void kernel_launch(void* const* d_in, const int* in_sizes, int n_in,
                              void* d_out, int out_size, void* d_ws, size_t ws_size,
                              hipStream_t stream) {
  (void)in_sizes; (void)n_in; (void)out_size; (void)ws_size;
  const float* X = (const float*)d_in[0];
  const float* E = (const float*)d_in[1];

  char* ws = (char*)d_ws;
  size_t off = 0;
  auto carve = [&](size_t bytes) -> void* {
    off = (off + 255) & ~(size_t)255;
    void* p = ws + off;
    off += bytes;
    return p;
  };
  float*  indeg  = (float*)carve(NN * 4);
  float*  outdeg = (float*)carve(NN * 4);
  float*  F      = (float*)carve(NN * 7 * 4);
  __bf16* Xin    = (__bf16*)carve(NN * 32 * 2);
  __bf16* t0     = (__bf16*)carve(NN * 256 * 2);
  __bf16* t1     = (__bf16*)carve(NN * 256 * 2);
  __bf16* hA     = (__bf16*)carve(NN * 256 * 2);
  __bf16* hB     = (__bf16*)carve(NN * 256 * 2);
  __bf16* Mbf    = (__bf16*)carve(NN * 256 * 2);
  float*  qf     = (float*)carve(NN * 256 * 4);
  float*  kf     = (float*)carve(NN * 256 * 4);
  float*  vf     = (float*)carve(NN * 256 * 4);
  float*  xo     = (float*)carve(NN * 16 * 4);
  float*  Lg     = (float*)carve((size_t)NN * NN * HH * 4);   // 8 MB
  __bf16* eA     = (__bf16*)carve((size_t)NN * NN * 64 * 2);  // 32 MB
  __bf16* eB     = (__bf16*)carve((size_t)NN * NN * 64 * 2);  // 32 MB

  struct PW { int wi, bi, din, dout, K, N; };
  static const PW tab[26] = {
    {3,2,4,256,32,256},   {5,4,256,256,256,256},  {7,6,256,64,256,64},      // in_E
    {9,8,15,256,32,256},  {11,10,256,256,256,256},{13,12,256,256,256,256},  // in_X
    {15,14,256,64,256,64},{17,16,256,256,256,256},{19,18,256,256,256,256},  // L0: Eout,K,Q
    {21,20,256,256,256,256},{23,22,256,256,256,256},                        // L0: V,Xout
    {25,24,64,256,64,256},{27,26,64,256,64,256},                            // L0: add,mul
    {29,28,256,64,256,64},{31,30,256,256,256,256},{33,32,256,256,256,256},  // L1: Eout,K,Q
    {35,34,256,256,256,256},{37,36,256,256,256,256},                        // L1: V,Xout
    {39,38,64,256,64,256},{41,40,64,256,64,256},                            // L1: add,mul
    {43,42,64,256,64,256},{45,44,256,256,256,256},{47,46,256,4,256,16},     // out_E
    {49,48,256,256,256,256},{51,50,256,256,256,256},{53,52,256,8,256,16},   // out_X
  };
  enum { INE0, INE1, INE2, INX0, INX1, INX2,
         L0_EOUT, L0_K, L0_Q, L0_V, L0_XOUT, L0_ADD, L0_MUL,
         L1_EOUT, L1_K, L1_Q, L1_V, L1_XOUT, L1_ADD, L1_MUL,
         OUTE0, OUTE1, OUTE2, OUTX0, OUTX1, OUTX2 };

  __bf16* Wp[26];
  float*  Bp[26];
  for (int p = 0; p < 26; ++p) {
    int K = tab[p].K, N = tab[p].N;
    Wp[p] = (__bf16*)carve((size_t)K * N * 2);
    Bp[p] = (float*)carve((size_t)N * 4);
    pack_w<<<(K * N + 255) / 256, 256, 0, stream>>>(
        (const float*)d_in[tab[p].wi], tab[p].dout, tab[p].din, Wp[p], K, N);
    pack_bias<<<(N + 255) / 256, 256, 0, stream>>>(
        (const float*)d_in[tab[p].bi], tab[p].dout, Bp[p], N);
  }

  auto gemm = [&](const __bf16* A, int M, int K, int p,
                  float* Cf, __bf16* Cb, int relu) {
    int N = tab[p].N;
    int ntiles = (M >> 4) * (N >> 4);
    gemm_node<<<(ntiles + 7) / 8, 256, 0, stream>>>(A, M, K, Wp[p], N, Bp[p],
                                                    Cf, Cb, relu);
  };

  // structural features + node embedding
  deg_kernel<<<NN, 256, 0, stream>>>(E, indeg, outdeg);
  featF_kernel<<<NN, 256, 0, stream>>>(E, indeg, outdeg, F);
  build_xin<<<(NN * 32 + 255) / 256, 256, 0, stream>>>(X, F, Xin);
  gemm(Xin, NN, 32,  INX0, nullptr, t0, 1);
  gemm(t0,  NN, 256, INX1, nullptr, t1, 1);
  gemm(t1,  NN, 256, INX2, nullptr, hA, 0);

  // edge embedding
  edge_in_mlp<<<(NN * NN) / 32, 256, 0, stream>>>(
      E, Wp[INE0], Bp[INE0], Wp[INE1], Bp[INE1], Wp[INE2], Bp[INE2], eA);

  __bf16* h = hA; __bf16* hn = hB;
  __bf16* ec = eA; __bf16* en = eB;
  for (int L = 0; L < 2; ++L) {
    int b = (L == 0) ? (int)L0_EOUT : (int)L1_EOUT;
    gemm(h, NN, 256, b + 2, qf, nullptr, 0);   // Q
    gemm(h, NN, 256, b + 1, kf, nullptr, 0);   // K
    gemm(h, NN, 256, b + 3, vf, nullptr, 0);   // V
    film_kernel<<<dim3(NN / 32, NN), 256, 0, stream>>>(
        ec, qf, kf, Wp[b + 5], Bp[b + 5], Wp[b + 6], Bp[b + 6],
        Wp[b + 0], Bp[b + 0], Lg, en);
    attn_kernel<<<NN, 256, 0, stream>>>(Lg, vf, Mbf);
    gemm(Mbf, NN, 256, b + 4, nullptr, hn, 0); // X_out
    { __bf16* tmp = h;  h = hn;  hn = tmp; }
    { __bf16* tmp = ec; ec = en; en = tmp; }
  }

  // output heads
  gemm(h,  NN, 256, OUTX0, nullptr, t0, 1);
  gemm(t0, NN, 256, OUTX1, nullptr, t1, 1);
  gemm(t1, NN, 256, OUTX2, xo, nullptr, 0);
  softmax_x<<<2, 256, 0, stream>>>(xo, (float*)d_out);

  edge_out_mlp<<<(NN * NN) / 32, 256, 0, stream>>>(
      ec, Wp[OUTE0], Bp[OUTE0], Wp[OUTE1], Bp[OUTE1], Wp[OUTE2], Bp[OUTE2],
      (float*)d_out + NN * 8);
}
